// RNN_43645457661967
// MI455X (gfx1250) — matrix operand order repaired and checked by reading, not compiled, not run on hardware
//
#include <hip/hip_runtime.h>

// ---------- types ----------
typedef __attribute__((ext_vector_type(16))) __bf16 v16bf;
typedef __attribute__((ext_vector_type(8)))  float  v8f;
typedef __attribute__((ext_vector_type(4)))  float  f32x4;
typedef int i32x4v __attribute__((vector_size(16)));   // gcc-style vector, matches builtin

union FragB16 { v16bf v; f32x4 q[2]; };

static constexpr int Tn = 100, Bsz = 256, NG = 2048, NP = 512;
static constexpr int LDS_STRIDE = 40;              // 32 k-elems + 8 pad (80B rows, 16B aligned)
static constexpr int LDS_BUF_ELEMS = 64 * LDS_STRIDE;

// CDNA5 async global->LDS path (guarded: falls back to ld+ds_store if absent)
#if defined(__gfx1250__) && __has_builtin(__builtin_amdgcn_global_load_async_to_lds_b128) && \
    __has_builtin(__builtin_amdgcn_s_wait_asynccnt)
#define HAVE_ASYNC_LDS 1
#else
#define HAVE_ASYNC_LDS 0
#endif

// ---------- deterministic noise (stand-in for jax.random.normal(key(1234))) ----------
__device__ __forceinline__ unsigned mix32(unsigned x) {
  x ^= x >> 16; x *= 0x7FEB352Du; x ^= x >> 15; x *= 0x846CA68Bu; x ^= x >> 16;
  return x;
}
__device__ __forceinline__ unsigned hash3(unsigned a, unsigned b, unsigned c) {
  return mix32(a * 0x9E3779B1u ^ mix32(b * 0x85EBCA77u ^ mix32(c * 0xC2B2AE3Du)));
}
__device__ __forceinline__ float nrm(unsigned a, unsigned b, unsigned c) {
  unsigned u1 = hash3(a, b, c);
  unsigned u2 = hash3(a ^ 0x6C62272Eu, b, c);
  float f1 = (float)((u1 >> 8) + 1u) * (1.0f / 16777216.0f);   // (0,1]
  float f2 = (float)(u2 >> 8) * (1.0f / 16777216.0f);          // [0,1)
  return sqrtf(-2.0f * __logf(f1)) * __cosf(6.28318530718f * f2);
}
__device__ __forceinline__ float noise_h(int t, int m, int n) {
  return nrm(0xA511E9B3u + (unsigned)t, (unsigned)m, (unsigned)n);
}
__device__ __forceinline__ float noise_v(int t, int m, int d) {
  return nrm(0xC2B2AE35u + (unsigned)t, (unsigned)m, (unsigned)d);
}

// ---------- WMMA fragment helpers (ISA 7.12.2 layouts, wave32) ----------
// A: 16x32 bf16. lanes 0-15: row m, K = {k..k+7, k+16..k+23}; lanes 16-31: +8.
__device__ __forceinline__ v16bf load_a_frag(const __bf16* __restrict__ A, int lda,
                                             int row, int kbase, int lane) {
  const int k0 = kbase + ((lane & 16) ? 8 : 0);
  const __bf16* p = A + (size_t)row * lda + k0;
  FragB16 f;
  f.q[0] = *(const f32x4*)(p);        // K = k0 .. k0+7
  f.q[1] = *(const f32x4*)(p + 16);   // K = k0+16 .. k0+23
  return f.v;
}
// B fragment read from an LDS-staged tile: rows are weight rows n (length-32 k chunk),
// lanes 0-15: K = {k..k+7, k+16..k+23} of column n; lanes 16-31: +8.
__device__ __forceinline__ v16bf load_b_lds(const __bf16* buf, int nl, int lane) {
  const int k0 = (lane & 16) ? 8 : 0;
  const __bf16* p = buf + nl * LDS_STRIDE + k0;
  FragB16 f;
  f.q[0] = *(const f32x4*)(p);
  f.q[1] = *(const f32x4*)(p + 16);
  return f.v;
}
__device__ __forceinline__ v8f wmma_bf16(v16bf a, v16bf b, v8f c) {
  return __builtin_amdgcn_wmma_f32_16x16x32_bf16(false, a, false, b, (short)0, c,
                                                 false, false);
}

// Stage one 64(n) x 32(k) bf16 B-tile into LDS: 256 threads, one b128 each.
__device__ __forceinline__ void stage_b_tile(const __bf16* __restrict__ W, int ldb,
                                             int n0, int kbase, int tid, __bf16* dstbuf) {
  const int nl = tid >> 2;             // 0..63
  const int kq = (tid & 3) * 8;        // 0,8,16,24
  const __bf16* src = W + (size_t)(n0 + nl) * ldb + kbase + kq;
  __bf16* dst = dstbuf + nl * LDS_STRIDE + kq;
#if HAVE_ASYNC_LDS
  __builtin_amdgcn_global_load_async_to_lds_b128(
      (__attribute__((address_space(1))) i32x4v*)const_cast<__bf16*>(src),
      (__attribute__((address_space(3))) i32x4v*)dst, 0, 0);
#else
  *(f32x4*)dst = *(const f32x4*)src;
#endif
}

// Block-cooperative GEMM tile: 8 waves x (16M x 64N), B shared via double-buffered LDS,
// A and next B tile prefetched while the current 4 WMMAs issue.
template <int KDIM>
__device__ __forceinline__ void mm_tile_block(const __bf16* __restrict__ A, int lda, int rowA,
                                              const __bf16* __restrict__ W, int ldb, int n0,
                                              int lane, int tid, __bf16* lds, v8f* acc) {
  constexpr int NT = KDIM / 32;
  const int lo = lane & 15;
  stage_b_tile(W, ldb, n0, 0, tid, lds);                      // tile 0 -> buf 0
  v16bf a_next = load_a_frag(A, lda, rowA, 0, lane);
  for (int kt = 0; kt < NT; ++kt) {
    __bf16* buf  = lds + (kt & 1) * LDS_BUF_ELEMS;
    __bf16* nbuf = lds + ((kt + 1) & 1) * LDS_BUF_ELEMS;
    const bool more = (kt + 1 < NT);
    if (more) stage_b_tile(W, ldb, n0, (kt + 1) * 32, tid, nbuf);   // prefetch next tile
#if HAVE_ASYNC_LDS
    if (more) __builtin_amdgcn_s_wait_asynccnt(1);   // current tile landed, next in flight
    else      __builtin_amdgcn_s_wait_asynccnt(0);
#endif
    __syncthreads();                                 // all waves' stages visible
    v16bf a = a_next;
    if (more) a_next = load_a_frag(A, lda, rowA, (kt + 1) * 32, lane);
#pragma unroll
    for (int j = 0; j < 4; ++j) {
      v16bf b = load_b_lds(buf, j * 16 + lo, lane);
      acc[j] = wmma_bf16(a, b, acc[j]);
    }
    __syncthreads();                                 // reads done before buf is overwritten
  }
}

// ---------- fp32 -> bf16 conversion ----------
__global__ void cvt_bf16_kernel(const float* __restrict__ x, __bf16* __restrict__ y, int n) {
  int i = blockIdx.x * blockDim.x + threadIdx.x;
  if (i < n) y[i] = (__bf16)x[i];
}

// ---------- h0 = p0 @ W_enc.T ; hp0 = bf16(h0 + 0.1*nh[0]) ----------
// grid 64 x 256thr: block tile 128M x 64N, K = NP = 512
__global__ __launch_bounds__(256) void h0_kernel(const __bf16* __restrict__ p0b,
                                                 const __bf16* __restrict__ Wencb,
                                                 __bf16* __restrict__ hp0) {
  __shared__ __bf16 ldsB[2 * LDS_BUF_ELEMS];
  const int tid = threadIdx.x, lane = tid & 31, wave = tid >> 5;
  const int m0 = (blockIdx.x >> 5) * 128 + wave * 16;
  const int n0 = (blockIdx.x & 31) * 64;
  const int lo = lane & 15, hi = (lane & 16) ? 8 : 0;

  v8f acc[4] = {};
  mm_tile_block<NP>(p0b, NP, m0 + lo, Wencb, NP, n0, lane, tid, ldsB, acc);

#pragma unroll
  for (int j = 0; j < 4; ++j) {
    const int n = n0 + j * 16 + lo;
#pragma unroll
    for (int i = 0; i < 8; ++i) {
      const int m = m0 + i + hi;
      hp0[(size_t)m * NG + n] = (__bf16)(acc[j][i] + 0.1f * noise_h(0, m, n));
    }
  }
}

// ---------- one recurrence step ----------
// g_t = relu(hp @ W_hid.T + vp @ W_vel.T) ; hp_next = bf16(g_t + 0.1*nh[t+1])
__global__ __launch_bounds__(256) void step_kernel(const __bf16* __restrict__ hp,
                                                   const __bf16* __restrict__ Whidb,
                                                   const float*  __restrict__ vt,
                                                   const float*  __restrict__ Wvel,
                                                   __bf16* __restrict__ g,
                                                   __bf16* __restrict__ hpn,
                                                   int t) {
  __shared__ __bf16 ldsB[2 * LDS_BUF_ELEMS];
  const int tid = threadIdx.x, lane = tid & 31, wave = tid >> 5;
  const int m0 = (blockIdx.x >> 5) * 128 + wave * 16;
  const int n0 = (blockIdx.x & 31) * 64;
  const int lo = lane & 15, hi = (lane & 16) ? 8 : 0;

  // velocity bias folded into accumulator init: c[m][n] = vp[m] . W_vel[n]
  float vp0[8], vp1[8];
#pragma unroll
  for (int i = 0; i < 8; ++i) {
    const int m = m0 + i + hi;
    vp0[i] = vt[(size_t)t * Bsz * 2 + m * 2 + 0] + 0.1f * noise_v(t, m, 0);
    vp1[i] = vt[(size_t)t * Bsz * 2 + m * 2 + 1] + 0.1f * noise_v(t, m, 1);
  }
  v8f acc[4];
#pragma unroll
  for (int j = 0; j < 4; ++j) {
    const int n = n0 + j * 16 + lo;
    const float w0 = Wvel[n * 2 + 0], w1 = Wvel[n * 2 + 1];
#pragma unroll
    for (int i = 0; i < 8; ++i) acc[j][i] = vp0[i] * w0 + vp1[i] * w1;
  }

  mm_tile_block<NG>(hp, NG, m0 + lo, Whidb, NG, n0, lane, tid, ldsB, acc);

#pragma unroll
  for (int j = 0; j < 4; ++j) {
    const int n = n0 + j * 16 + lo;
#pragma unroll
    for (int i = 0; i < 8; ++i) {
      const int m = m0 + i + hi;
      const float h = fmaxf(acc[j][i], 0.0f);
      g[(size_t)m * NG + n]   = (__bf16)h;
      hpn[(size_t)m * NG + n] = (__bf16)(h + 0.1f * noise_h(t + 1, m, n));
    }
  }
}

// ---------- decoder: out_t = g_t @ W_dec.T  (f32 out) ----------
// grid 16 x 256thr: block tile 128M x 64N, K = NG, N = NP = 512
__global__ __launch_bounds__(256) void dec_kernel(const __bf16* __restrict__ g,
                                                  const __bf16* __restrict__ Wdecb,
                                                  float* __restrict__ out) {
  __shared__ __bf16 ldsB[2 * LDS_BUF_ELEMS];
  const int tid = threadIdx.x, lane = tid & 31, wave = tid >> 5;
  const int m0 = (blockIdx.x >> 3) * 128 + wave * 16;
  const int n0 = (blockIdx.x & 7) * 64;
  const int lo = lane & 15, hi = (lane & 16) ? 8 : 0;

  v8f acc[4] = {};
  mm_tile_block<NG>(g, NG, m0 + lo, Wdecb, NG, n0, lane, tid, ldsB, acc);

#pragma unroll
  for (int j = 0; j < 4; ++j) {
    const int n = n0 + j * 16 + lo;
#pragma unroll
    for (int i = 0; i < 8; ++i) {
      const int m = m0 + i + hi;
      out[(size_t)m * NP + n] = acc[j][i];
    }
  }
}

// ---------- host ----------
extern "C" void kernel_launch(void* const* d_in, const int* in_sizes, int n_in,
                              void* d_out, int out_size, void* d_ws, size_t ws_size,
                              hipStream_t stream) {
  const float* vt   = (const float*)d_in[0];  // [T,B,2]
  const float* p0   = (const float*)d_in[1];  // [B,NP]
  const float* Wenc = (const float*)d_in[2];  // [NG,NP]
  const float* Wvel = (const float*)d_in[3];  // [NG,2]
  const float* Whid = (const float*)d_in[4];  // [NG,NG]
  const float* Wdec = (const float*)d_in[5];  // [NP,NG]
  float* out = (float*)d_out;                 // [T,B,NP]

  char* ws = (char*)d_ws;
  __bf16* Wencb = (__bf16*)ws; ws += (size_t)NG * NP * 2;
  __bf16* Whidb = (__bf16*)ws; ws += (size_t)NG * NG * 2;
  __bf16* Wdecb = (__bf16*)ws; ws += (size_t)NP * NG * 2;
  __bf16* p0b   = (__bf16*)ws; ws += (size_t)Bsz * NP * 2;
  __bf16* hpA   = (__bf16*)ws; ws += (size_t)Bsz * NG * 2;
  __bf16* hpB   = (__bf16*)ws; ws += (size_t)Bsz * NG * 2;
  __bf16* gbuf  = (__bf16*)ws; ws += (size_t)Bsz * NG * 2;

  // per-call bf16 conversions (weights are L2-resident thereafter)
  {
    int n;
    n = NG * NP;   cvt_bf16_kernel<<<(n + 255) / 256, 256, 0, stream>>>(Wenc, Wencb, n);
    n = NG * NG;   cvt_bf16_kernel<<<(n + 255) / 256, 256, 0, stream>>>(Whid, Whidb, n);
    n = NP * NG;   cvt_bf16_kernel<<<(n + 255) / 256, 256, 0, stream>>>(Wdec, Wdecb, n);
    n = Bsz * NP;  cvt_bf16_kernel<<<(n + 255) / 256, 256, 0, stream>>>(p0, p0b, n);
  }

  h0_kernel<<<64, 256, 0, stream>>>(p0b, Wencb, hpA);

  __bf16* hp[2] = {hpA, hpB};
  for (int t = 0; t < Tn; ++t) {
    step_kernel<<<64, 256, 0, stream>>>(hp[t & 1], Whidb, vt, Wvel, gbuf,
                                        hp[(t + 1) & 1], t);
    dec_kernel<<<16, 256, 0, stream>>>(gbuf, Wdecb, out + (size_t)t * Bsz * NP);
  }
}
